// TrajEncoder_13108240187766
// MI455X (gfx1250) — matrix operand order repaired and checked by reading, not compiled, not run on hardware
//
#include <hip/hip_runtime.h>
#include <hip/hip_bf16.h>

// ---------------------------------------------------------------------------
// TrajEncoder (embedding gather + LSTM scan + last-valid gather) for MI455X.
//
// Shapes: VOCAB=100000, D=128, H=256 (4H=1024), B=64, T=512.
//
//  * 4 workgroups, each owns 16 batch rows (one WMMA M-tile) and runs the
//    whole T=512 scan independently (batch rows never interact).
//  * 16 waves / WG.  Wave w owns gate tiles nt = {w, w+16, w+32, w+48} =
//    columns j=w*16..w*16+15 of gates i,f,g,o -> the LSTM elementwise update
//    happens entirely in registers (c-state lives in VGPRs for all 512 steps).
//  * Matmuls via v_wmma_f32_16x16x32_bf16 (f32 accumulate).  Weights are
//    pre-packed into the native wave32 B-operand layout so B loads are
//    coalesced 32B/lane global_load_b128 pairs that stay hot in L2 (768 KB).
//  * The weight addresses get an opaque per-iteration integer offset
//    (asm-blinded, value 0) so the compiler can neither hoist 384 VGPRs of
//    loop-invariant weights out of the t-loop (round-1 pathology: scratch
//    spill reloads) nor does it lose global address-space inference
//    (round-2 pathology: "+s"-blinded *pointers* degraded to flat_load,
//    which ticks DScnt and entangles with the LDS h reads).
//  * Input projection fused into the scan (K chunks 0..3 from the gathered
//    embedding row, f32->bf16 in registers), so workspace is <1 MB.
//  * h staged per-step in LDS as bf16, padded row stride 264 -> the
//    A-operand reads (16B per lane) are bank-conflict free.
// ---------------------------------------------------------------------------

typedef __bf16 bf16_t;
typedef __attribute__((ext_vector_type(16))) __bf16 v16bf;
typedef __attribute__((ext_vector_type(8)))  __bf16 v8bf;
typedef __attribute__((ext_vector_type(8)))  float  v8f;

#define TE_D    128     // embedding dim (K for input projection)
#define TE_H    256     // hidden dim
#define TE_G4   1024    // 4*H
#define TE_B    64
#define TE_T    512
#define TE_HSTR 264     // padded LDS row stride for h (bf16), conflict-free

#define NT_IH   (4 * 64 * 32 * 16)   // packed W_ih elements (kc=4 chunks)
#define NT_HH   (8 * 64 * 32 * 16)   // packed W_hh elements (kc=8 chunks)

// --- Pack W_ih / W_hh (f32, row-major [4H, K]) into wave32 WMMA B-operand
// --- layout, bf16:  pk[((kc*64 + nt)*32 + lane)*16 + e] = W[N][K]
// --- with N = nt*16 + (lane&15),  K = kc*32 + 8*(lane>>4) + (e&7) + 16*(e>>3).
__global__ void traj_pack_weights(const float* __restrict__ W_ih,
                                  const float* __restrict__ W_hh,
                                  bf16_t* __restrict__ pk_ih,
                                  bf16_t* __restrict__ pk_hh) {
  int idx = blockIdx.x * blockDim.x + threadIdx.x;
  if (idx < NT_IH) {
    int e = idx & 15, lane = (idx >> 4) & 31, nt = (idx >> 9) & 63, kc = idx >> 15;
    int K = kc * 32 + 8 * (lane >> 4) + (e & 7) + 16 * (e >> 3);
    int N = nt * 16 + (lane & 15);
    pk_ih[idx] = (bf16_t)W_ih[N * TE_D + K];
  } else if (idx < NT_IH + NT_HH) {
    int i2 = idx - NT_IH;
    int e = i2 & 15, lane = (i2 >> 4) & 31, nt = (i2 >> 9) & 63, kc = i2 >> 15;
    int K = kc * 32 + 8 * (lane >> 4) + (e & 7) + 16 * (e >> 3);
    int N = nt * 16 + (lane & 15);
    pk_hh[i2] = (bf16_t)W_hh[N * TE_H + K];
  }
}

__global__ __launch_bounds__(512, 1)
void traj_lstm_scan(const int*   __restrict__ path,        // [B, T]
                    const int*   __restrict__ valid_len,   // [B]
                    const float* __restrict__ embedding,   // [VOCAB, D]
                    const float* __restrict__ b_ih,        // [4H]
                    const float* __restrict__ b_hh,        // [4H]
                    const bf16_t* __restrict__ pk_ih,      // packed, 4 chunks
                    const bf16_t* __restrict__ pk_hh,      // packed, 8 chunks
                    float* __restrict__ out)               // [B, H]
{
  __shared__ __align__(16) bf16_t hS[16][TE_HSTR];   // h (bf16), padded rows

  const int tid   = threadIdx.x;
  const int lane  = tid & 31;
  const int wave  = tid >> 5;          // 0..15
  const int half  = lane >> 4;         // 0/1
  const int mrow  = lane & 15;         // M index this lane supplies for A
  const int bBase = blockIdx.x * 16;   // this WG's batch block
  const int j     = wave * 16 + mrow;  // hidden column this lane owns (0..255)

  // Per-lane combined biases for the 4 gates at column j (pytorch i,f,g,o).
  const float bji = b_ih[j]       + b_hh[j];
  const float bjf = b_ih[j + 256] + b_hh[j + 256];
  const float bjg = b_ih[j + 512] + b_hh[j + 512];
  const float bjo = b_ih[j + 768] + b_hh[j + 768];

  // valid_len for the 8 batch rows this lane's accumulator elements cover
  // (C layout: element v -> m = v + 8*half).
  int vlr[8];
  #pragma unroll
  for (int v = 0; v < 8; ++v) vlr[v] = valid_len[bBase + v + 8 * half];

  float cst[8];                        // persistent cell state (registers)
  #pragma unroll
  for (int v = 0; v < 8; ++v) cst[v] = 0.f;

  // zero h staging buffer
  for (int i = tid; i < 16 * TE_HSTR; i += 512) ((bf16_t*)hS)[i] = (bf16_t)0.f;
  __syncthreads();

  // Per-wave base offset into the packed weights (elements); nt = wave+16*i.
  const size_t wBase = ((size_t)wave * 32 + lane) * 16;

  for (int t = 0; t < TE_T; ++t) {
    // Opaque zero offset, renewed every iteration: keeps the B-operand loads
    // inside the loop (no hoist -> no spill) while preserving the global
    // address space of pk_ih / pk_hh (stays global_load, LOADcnt only).
    size_t blind = 0;
    asm volatile("" : "+s"(blind));

    v8f acc[4];                        // acc[0..3] = i,f,g,o gate tiles
    #pragma unroll
    for (int i = 0; i < 4; ++i)
      #pragma unroll
      for (int v = 0; v < 8; ++v) acc[i][v] = 0.f;

    // gathered embedding row for the M-row this lane supplies
    const float* xrow =
        embedding + (size_t)path[(size_t)(bBase + mrow) * TE_T + t] * TE_D;

    // prefetch next step's row (hide gather latency behind WMMA work)
    if (t + 1 < TE_T) {
      const float* nrow =
          embedding + (size_t)path[(size_t)(bBase + mrow) * TE_T + t + 1] * TE_D;
      __builtin_prefetch(nrow + half * 64, 0, 3);
      __builtin_prefetch(nrow + half * 64 + 32, 0, 3);
    }

    // ---- input projection: gates += x @ W_ih^T  (K = 128, 4 chunks) ----
    #pragma unroll
    for (int kc = 0; kc < 4; ++kc) {
      // A layout (16-bit, 16x32): half=0 -> K = kc*32 + {0..7, 16..23}
      //                           half=1 -> K = kc*32 + {8..15, 24..31}
      const float* p0 = xrow + kc * 32 + half * 8;
      float4 x0 = *(const float4*)(p0);
      float4 x1 = *(const float4*)(p0 + 4);
      float4 x2 = *(const float4*)(p0 + 16);
      float4 x3 = *(const float4*)(p0 + 20);
      v16bf a;
      a[0]=(bf16_t)x0.x; a[1]=(bf16_t)x0.y; a[2]=(bf16_t)x0.z; a[3]=(bf16_t)x0.w;
      a[4]=(bf16_t)x1.x; a[5]=(bf16_t)x1.y; a[6]=(bf16_t)x1.z; a[7]=(bf16_t)x1.w;
      a[8]=(bf16_t)x2.x; a[9]=(bf16_t)x2.y; a[10]=(bf16_t)x2.z; a[11]=(bf16_t)x2.w;
      a[12]=(bf16_t)x3.x; a[13]=(bf16_t)x3.y; a[14]=(bf16_t)x3.z; a[15]=(bf16_t)x3.w;

      // Load all 4 B tiles first so the loads overlap, then issue WMMAs.
      v16bf bm[4];
      #pragma unroll
      for (int i = 0; i < 4; ++i)
        bm[i] = *(const v16bf*)
            (pk_ih + blind + wBase + ((size_t)kc * 64 + 16 * i) * 32 * 16);
      #pragma unroll
      for (int i = 0; i < 4; ++i)
        acc[i] = __builtin_amdgcn_wmma_f32_16x16x32_bf16(
            false, a, false, bm[i], (short)0, acc[i], false, false);
    }

    // ---- recurrence: gates += h @ W_hh^T  (K = 256, 8 chunks) ----
    #pragma unroll
    for (int kc = 0; kc < 8; ++kc) {
      const v8bf lo = *(const v8bf*)&hS[mrow][kc * 32 + half * 8];
      const v8bf hi = *(const v8bf*)&hS[mrow][kc * 32 + 16 + half * 8];
      v16bf a;
      #pragma unroll
      for (int e = 0; e < 8; ++e) { a[e] = lo[e]; a[e + 8] = hi[e]; }

      v16bf bm[4];
      #pragma unroll
      for (int i = 0; i < 4; ++i)
        bm[i] = *(const v16bf*)
            (pk_hh + blind + wBase + ((size_t)kc * 64 + 16 * i) * 32 * 16);
      #pragma unroll
      for (int i = 0; i < 4; ++i)
        acc[i] = __builtin_amdgcn_wmma_f32_16x16x32_bf16(
            false, a, false, bm[i], (short)0, acc[i], false, false);
    }

    __syncthreads();   // everyone done reading hS before it is overwritten

    // ---- elementwise LSTM cell, fully in registers ----
    // C layout: acc[g][v] is gate g at (m = v + 8*half, column j).
    #pragma unroll
    for (int v = 0; v < 8; ++v) {
      float gi = acc[0][v] + bji;
      float gf = acc[1][v] + bjf;
      float gg = acc[2][v] + bjg;
      float go = acc[3][v] + bjo;
      float is = 1.f / (1.f + __expf(-gi));
      float fs = 1.f / (1.f + __expf(-gf));
      float os = 1.f / (1.f + __expf(-go));
      float c  = fs * cst[v] + is * tanhf(gg);
      cst[v]   = c;
      float h  = os * tanhf(c);
      hS[v + 8 * half][j] = (bf16_t)h;
      if (t == vlr[v] - 1)
        out[(size_t)(bBase + v + 8 * half) * TE_H + j] = h;
    }

    __syncthreads();   // new h visible before next step's A reads
  }
}

extern "C" void kernel_launch(void* const* d_in, const int* in_sizes, int n_in,
                              void* d_out, int out_size, void* d_ws, size_t ws_size,
                              hipStream_t stream) {
  const int*   path      = (const int*)d_in[0];
  const int*   valid_len = (const int*)d_in[1];
  const float* embedding = (const float*)d_in[2];
  const float* W_ih      = (const float*)d_in[3];
  const float* W_hh      = (const float*)d_in[4];
  const float* b_ih      = (const float*)d_in[5];
  const float* b_hh      = (const float*)d_in[6];
  float* out = (float*)d_out;

  bf16_t* pk_ih = (bf16_t*)d_ws;
  bf16_t* pk_hh = (bf16_t*)((char*)d_ws + (size_t)NT_IH * sizeof(bf16_t));

  const int total = NT_IH + NT_HH;
  traj_pack_weights<<<(total + 255) / 256, 256, 0, stream>>>(
      W_ih, W_hh, pk_ih, pk_hh);

  traj_lstm_scan<<<TE_B / 16, 512, 0, stream>>>(
      path, valid_len, embedding, b_ih, b_hh, pk_ih, pk_hh, out);
}
